// Refiner_8083128451217
// MI455X (gfx1250) — matrix operand order repaired and checked by reading, not compiled, not run on hardware
//
#include <hip/hip_runtime.h>

// ---------------------------------------------------------------------------
// MI455X (gfx1250, wave32) implementation of the attention "Refiner".
// bf16 WMMA (v_wmma_f32_16x16x32_bf16) for all GEMM-like work; fused
// flash-style attention; K/V tiles staged into LDS with CDNA5
// GLOBAL_LOAD_ASYNC_TO_LDS_B128 (ASYNCcnt) — layouts arranged so the staging
// is a straight block copy and every WMMA fragment read is a 16B LDS read.
// ---------------------------------------------------------------------------

typedef __bf16 bf16;
typedef __attribute__((ext_vector_type(16))) __bf16 v16bf;
typedef __attribute__((ext_vector_type(8)))  float  v8f;

#define B_      2
#define HEADS_  4
#define DHEAD   128
#define CH      128
#define N_      2304
#define INNER   512

struct alignas(16) U128 { unsigned int x, y, z, w; };
union Frag { v16bf v; U128 u4[2]; __bf16 h[16]; };   // 32B: one 16-bit A/B fragment

__device__ __forceinline__ v8f wmma_bf16(v16bf a, v16bf b, v8f c) {
  // D = A(16x32) * B(32x16) + C, f32 accumulate
  return __builtin_amdgcn_wmma_f32_16x16x32_bf16(false, a, false, b, (short)0, c, false, false);
}

// CDNA5 async copy: 16 bytes per lane, global -> LDS, tracked by ASYNCcnt.
__device__ __forceinline__ void async_load_b128(unsigned lds_off, const void* gptr) {
  asm volatile("global_load_async_to_lds_b128 %0, %1, off"
               :: "v"(lds_off), "v"(gptr) : "memory");
}
__device__ __forceinline__ void wait_async0() {
  asm volatile("s_wait_asynccnt 0x0" ::: "memory");
}
__device__ __forceinline__ unsigned lds_off32(const void* p) {
  // generic pointers to LDS carry the byte offset in the low 32 bits
  return (unsigned)(unsigned long long)p;
}

// ---------------------------------------------------------------------------
// 3x3 stride-2 pad-1 conv (flow pyramid). <10% of total FLOPs -> plain VALU.
// ---------------------------------------------------------------------------
__global__ void conv3x3s2_kernel(const float* __restrict__ x, const float* __restrict__ W,
                                 const float* __restrict__ bias, float* __restrict__ y,
                                 int Cin, int Cout, int Hin, int Win)
{
  const int Hout = Hin >> 1, Wout = Win >> 1;
  const size_t total = (size_t)B_ * Cout * Hout * Wout;
  size_t idx = (size_t)blockIdx.x * blockDim.x + threadIdx.x;
  if (idx >= total) return;
  const int ox = (int)(idx % Wout);
  size_t t = idx / Wout;
  const int oy = (int)(t % Hout); t /= Hout;
  const int oc = (int)(t % Cout);
  const int bb = (int)(t / Cout);
  float acc = bias[oc];
  const int iy0 = 2 * oy - 1, ix0 = 2 * ox - 1;
  for (int ic = 0; ic < Cin; ++ic) {
    const float* xp = x + ((size_t)bb * Cin + ic) * Hin * Win;
    const float* wp = W + ((size_t)oc * Cin + ic) * 9;
#pragma unroll
    for (int ky = 0; ky < 3; ++ky) {
      const int iy = iy0 + ky;
      if (iy < 0 || iy >= Hin) continue;
#pragma unroll
      for (int kx = 0; kx < 3; ++kx) {
        const int ix = ix0 + kx;
        if (ix < 0 || ix >= Win) continue;
        acc = fmaf(xp[(size_t)iy * Win + ix], wp[ky * 3 + kx], acc);
      }
    }
  }
  y[idx] = acc;
}

// ---------------------------------------------------------------------------
// QKV projection: out[o, (b,n)] = sum_c W[o,c] * X[b,c,n], via bf16 WMMA.
// One wave computes a 16(o) x 16(n) tile, K-loop over C=128 in 4 chunks of 32.
// layout 0: store [b][h][n][dd]  (Q with SCALE, and K -> row-major [m][d])
// layout 1: store [b][h][dd][m]  (the three V streams -> d-major [d][m])
// Layouts are chosen so the attention kernel's LDS tiles are straight copies.
// ---------------------------------------------------------------------------
__global__ __launch_bounds__(128) void proj_qkv_kernel(
    const float* __restrict__ W, const float* __restrict__ X,
    bf16* __restrict__ out, int layout, float scale)
{
  const int tile = blockIdx.x * 4 + (threadIdx.x >> 5);
  const int ot   = tile / 288;          // 32 o-tiles
  const int ct   = tile % 288;          // 288 col-tiles (B*N/16)
  const int o0   = ot * 16;
  const int col0 = ct * 16;
  const int b    = col0 / N_;
  const int n0   = col0 % N_;
  const int lane = threadIdx.x & 31, grp = lane >> 4, lg = lane & 15;

  v8f acc = {0.f, 0.f, 0.f, 0.f, 0.f, 0.f, 0.f, 0.f};
#pragma unroll
  for (int c0 = 0; c0 < CH; c0 += 32) {
    Frag a, bb;
    // A fragment: W rows are contiguous in c  (lane: row o=o0+lg, K runs 16*grp..+15)
    const float* wrow = W + (size_t)(o0 + lg) * CH + c0 + 16 * grp;
#pragma unroll
    for (int i = 0; i < 8; ++i) { a.h[i] = (__bf16)wrow[i]; a.h[8 + i] = (__bf16)wrow[8 + i]; }
    // B fragment: X[b][c][n], lane: col n=n0+lg, rows c = c0+16*grp..+15 (stride N)
    const float* xcol = X + ((size_t)b * CH + c0 + 16 * grp) * N_ + n0 + lg;
#pragma unroll
    for (int i = 0; i < 16; ++i) bb.h[i] = (__bf16)xcol[(size_t)i * N_];
    acc = wmma_bf16(a.v, bb.v, acc);
  }
#pragma unroll
  for (int r = 0; r < 8; ++r) {
    const int oo = o0 + r + 8 * grp;          // D row (o), lane col (n)
    const int hh = oo >> 7, dd = oo & 127;
    const int n  = n0 + lg;
    const size_t bh = (size_t)b * HEADS_ + hh;
    const float val = acc[r] * scale;
    if (layout == 0) out[(bh * N_ + n) * DHEAD + dd] = (__bf16)val;
    else             out[(bh * DHEAD + dd) * N_ + n] = (__bf16)val;
  }
}

// ---------------------------------------------------------------------------
// Fused flash attention, 3 value streams sharing one softmax.
// Block = 256 thr (8 waves); each wave owns 16 query rows; block covers 128.
// Per 32-wide M tile:
//   - async-copy K tile (k[m][d], 32x128) and 3 V tiles (v[d][m], 128x32)
//     straight into LDS via global_load_async_to_lds_b128, s_wait_asynccnt,
//   - 8 WMMAs for S, online softmax (shfl reductions + v_exp_f32),
//   - P -> LDS -> A-fragment, 24 WMMAs for the 3 P*V streams.
// Output: att[s][b][n][h*128+dd] bf16, normalized.
// ---------------------------------------------------------------------------
__global__ __launch_bounds__(256) void attn_kernel(
    const bf16* __restrict__ q, const bf16* __restrict__ k,
    const bf16* __restrict__ v, bf16* __restrict__ att)
{
  __shared__ alignas(16) __bf16 sK[32][128];       //  8KB  [m][dd]  (copy of k tile)
  __shared__ alignas(16) __bf16 sV[3][128][32];    // 24KB  [s][dd][m] (copy of v tiles)
  __shared__ alignas(16) __bf16 sP[8][16][32];     //  8KB  per-wave P tile

  const int bh    = blockIdx.x / 18;
  const int strip = blockIdx.x % 18;
  const int b = bh >> 2;
  const int h = bh & 3;
  const int tid  = threadIdx.x;
  const int wave = tid >> 5;
  const int lane = tid & 31;
  const int grp  = lane >> 4;
  const int lg   = lane & 15;
  const int n0   = strip * 128 + wave * 16;

  // Q A-fragments (rows contiguous in d; loaded once)
  const bf16* qrow = q + ((size_t)bh * N_ + n0 + lg) * DHEAD;
  Frag qa[4];
#pragma unroll
  for (int d = 0; d < 4; ++d) {
    qa[d].u4[0] = *(const U128*)(qrow + d * 32 + 16 * grp);
    qa[d].u4[1] = *(const U128*)(qrow + d * 32 + 16 * grp + 8);
  }

  const v8f vzero = {0.f, 0.f, 0.f, 0.f, 0.f, 0.f, 0.f, 0.f};
  v8f o[3][8];
#pragma unroll
  for (int s = 0; s < 3; ++s)
#pragma unroll
    for (int d8 = 0; d8 < 8; ++d8) o[s][d8] = vzero;

  float mrow[8], lrow[8];
#pragma unroll
  for (int r = 0; r < 8; ++r) { mrow[r] = -1e30f; lrow[r] = 0.0f; }

  const bf16* kbh = k + (size_t)bh * N_ * DHEAD;   // k: [bh][m][dd]

  for (int m0 = 0; m0 < N_; m0 += 32) {
    __syncthreads();   // previous tile fully consumed before overwrite

    // --- async staging: 2048 16B chunks (8 per thread), straight copies ---
    for (int u = tid; u < 2048; u += 256) {
      if (u < 512) {                       // K tile: 32 rows x 256B
        const int mm = u >> 4, part = u & 15;
        const bf16* g = kbh + (size_t)(m0 + mm) * DHEAD + part * 8;
        async_load_b128(lds_off32(&sK[mm][part * 8]), g);
        if (m0 + 32 < N_) __builtin_prefetch(g + 32 * DHEAD, 0, 1);
      } else {                             // V tiles: 3 x 128 rows x 64B
        const int c = u - 512;
        const int s = c >> 9, rem = c & 511, dd = rem >> 2, part = rem & 3;
        const bf16* g = v + (((size_t)s * 8 + bh) * DHEAD + dd) * N_ + m0 + part * 8;
        async_load_b128(lds_off32(&sV[s][dd][part * 8]), g);
      }
    }
    wait_async0();       // s_wait_asynccnt 0: this wave's copies landed
    __syncthreads();     // all waves' copies visible

    // --- S = Q * K^T for a 16x32 tile (two 16x16 accumulators) ---
    v8f c0 = vzero, c1 = vzero;
#pragma unroll
    for (int d = 0; d < 4; ++d) {
      Frag kb0, kb1;   // B frag: lane col = m, rows dd contiguous in sK row
      kb0.u4[0] = *(const U128*)&sK[lg][d * 32 + 16 * grp];
      kb0.u4[1] = *(const U128*)&sK[lg][d * 32 + 16 * grp + 8];
      kb1.u4[0] = *(const U128*)&sK[16 + lg][d * 32 + 16 * grp];
      kb1.u4[1] = *(const U128*)&sK[16 + lg][d * 32 + 16 * grp + 8];
      c0 = wmma_bf16(qa[d].v, kb0.v, c0);
      c1 = wmma_bf16(qa[d].v, kb1.v, c1);
    }

    // --- online softmax: rows live as (r + 8*grp); reduce across 16-lane group ---
    float alpha[8];
#pragma unroll
    for (int r = 0; r < 8; ++r) {
      float mx = fmaxf(c0[r], c1[r]);
#pragma unroll
      for (int off = 1; off <= 8; off <<= 1) mx = fmaxf(mx, __shfl_xor(mx, off));
      const float mnew = fmaxf(mrow[r], mx);
      alpha[r] = __expf(mrow[r] - mnew);
      mrow[r]  = mnew;
      const float p0 = __expf(c0[r] - mnew);
      const float p1 = __expf(c1[r] - mnew);
      c0[r] = p0; c1[r] = p1;
      float rs = p0 + p1;
#pragma unroll
      for (int off = 1; off <= 8; off <<= 1) rs += __shfl_xor(rs, off);
      lrow[r] = lrow[r] * alpha[r] + rs;
    }
#pragma unroll
    for (int s = 0; s < 3; ++s)
#pragma unroll
      for (int d8 = 0; d8 < 8; ++d8)
#pragma unroll
        for (int r = 0; r < 8; ++r) o[s][d8][r] *= alpha[r];

    // --- P (C layout, f32) -> LDS bf16 -> A-fragment layout ---
#pragma unroll
    for (int r = 0; r < 8; ++r) {
      sP[wave][r + 8 * grp][lg]      = (__bf16)c0[r];
      sP[wave][r + 8 * grp][16 + lg] = (__bf16)c1[r];
    }
    __syncthreads();
    Frag pa;
    pa.u4[0] = *(const U128*)&sP[wave][lg][16 * grp];
    pa.u4[1] = *(const U128*)&sP[wave][lg][16 * grp + 8];

    // --- O_s += P * V_s  (K-dim = full 32-wide m tile, 8 d-chunks of 16) ---
#pragma unroll
    for (int s = 0; s < 3; ++s)
#pragma unroll
      for (int d8 = 0; d8 < 8; ++d8) {
        Frag vb;   // lane col = dd, rows mm contiguous in sV row
        vb.u4[0] = *(const U128*)&sV[s][d8 * 16 + lg][16 * grp];
        vb.u4[1] = *(const U128*)&sV[s][d8 * 16 + lg][16 * grp + 8];
        o[s][d8] = wmma_bf16(pa.v, vb.v, o[s][d8]);
      }
  }

  // --- normalize and emit att[s][b][n][h*128+dd] (bf16) ---
#pragma unroll
  for (int s = 0; s < 3; ++s)
#pragma unroll
    for (int d8 = 0; d8 < 8; ++d8) {
      const int ci = h * DHEAD + d8 * 16 + lg;
#pragma unroll
      for (int r = 0; r < 8; ++r) {
        const int n = n0 + r + 8 * grp;
        att[(((size_t)s * B_ + b) * N_ + n) * INNER + ci] =
            (__bf16)(o[s][d8][r] / lrow[r]);
      }
    }
}

// ---------------------------------------------------------------------------
// Output projection + residual: out[b,co,n] = res + gamma * sum_ci Wp[co,ci]*att[b,n,ci]
// D tile = 16(n) x 16(co); A = att rows (contiguous), B = Wp rows (contiguous).
// ---------------------------------------------------------------------------
__global__ __launch_bounds__(128) void proj_out_kernel(
    const bf16* __restrict__ att, const float* __restrict__ Wp,
    const float* __restrict__ res, const float* __restrict__ gamma,
    float* __restrict__ out)
{
  const int tile = blockIdx.x * 4 + (threadIdx.x >> 5);
  const int b    = tile / (144 * 8);
  const int rem  = tile % (144 * 8);
  const int nt   = rem >> 3;
  const int cot  = rem & 7;
  const int n0 = nt * 16, co0 = cot * 16;
  const int lane = threadIdx.x & 31, grp = lane >> 4, lg = lane & 15;

  v8f acc = {0.f, 0.f, 0.f, 0.f, 0.f, 0.f, 0.f, 0.f};
  const bf16*  arow = att + ((size_t)b * N_ + n0 + lg) * INNER;
  const float* wrow = Wp + (size_t)(co0 + lg) * INNER;
#pragma unroll
  for (int ci0 = 0; ci0 < INNER; ci0 += 32) {
    Frag a, bb;
    a.u4[0] = *(const U128*)(arow + ci0 + 16 * grp);
    a.u4[1] = *(const U128*)(arow + ci0 + 16 * grp + 8);
#pragma unroll
    for (int i = 0; i < 16; ++i) bb.h[i] = (__bf16)wrow[ci0 + 16 * grp + i];
    acc = wmma_bf16(a.v, bb.v, acc);
  }
  const float g  = *gamma;
  const int   co = co0 + lg;
#pragma unroll
  for (int r = 0; r < 8; ++r) {
    const int n = n0 + r + 8 * grp;
    const size_t idx = ((size_t)b * CH + co) * N_ + n;
    out[idx] = res[idx] + g * acc[r];
  }
}

// ---------------------------------------------------------------------------
extern "C" void kernel_launch(void* const* d_in, const int* in_sizes, int n_in,
                              void* d_out, int out_size, void* d_ws, size_t ws_size,
                              hipStream_t stream)
{
  const float* prev_frame = (const float*)d_in[0];
  const float* curr_frame = (const float*)d_in[1];
  const float* prev_flow  = (const float*)d_in[2];
  const float* prev_net   = (const float*)d_in[3];
  const float* prev_inp   = (const float*)d_in[4];
  const float* Wq  = (const float*)d_in[5];
  const float* Wk  = (const float*)d_in[6];
  const float* Wvf = (const float*)d_in[7];
  const float* Wvn = (const float*)d_in[8];
  const float* Wvi = (const float*)d_in[9];
  const float* Wr1 = (const float*)d_in[10]; const float* br1 = (const float*)d_in[11];
  const float* Wr2 = (const float*)d_in[12]; const float* br2 = (const float*)d_in[13];
  const float* Wr3 = (const float*)d_in[14]; const float* br3 = (const float*)d_in[15];
  const float* Wpf = (const float*)d_in[16];
  const float* Wpn = (const float*)d_in[17];
  const float* Wpi = (const float*)d_in[18];
  const float* gamma_flow = (const float*)d_in[19];
  const float* gamma_net  = (const float*)d_in[20];
  const float* gamma_inp  = (const float*)d_in[21];
  (void)in_sizes; (void)n_in; (void)out_size; (void)ws_size;

  // workspace carve (~68 MB total)
  char* w = (char*)d_ws;
  auto carve = [&](size_t bytes) -> char* {
    char* p = w; w += (bytes + 255) & ~(size_t)255; return p;
  };
  float* c1   = (float*)carve((size_t)2 * 64 * 192 * 192 * 4);
  float* c2   = (float*)carve((size_t)2 * 128 * 96 * 96 * 4);
  float* pf   = (float*)carve((size_t)2 * 128 * 48 * 48 * 4);
  bf16*  qb   = (bf16*)carve((size_t)2 * 4 * N_ * DHEAD * 2);
  bf16*  kb   = (bf16*)carve((size_t)2 * 4 * N_ * DHEAD * 2);
  bf16*  vb   = (bf16*)carve((size_t)3 * 2 * 4 * N_ * DHEAD * 2);
  bf16*  attb = (bf16*)carve((size_t)3 * 2 * N_ * INNER * 2);

  const float SCALE = 0.08838834764831845f;  // 128^-0.5

  // flow pyramid: [2,2,384,384] -> [2,64,192,192] -> [2,128,96,96] -> [2,128,48,48]
  conv3x3s2_kernel<<<(2 * 64 * 192 * 192) / 256, 256, 0, stream>>>(prev_flow, Wr1, br1, c1, 2, 64, 384, 384);
  conv3x3s2_kernel<<<(2 * 128 * 96 * 96) / 256, 256, 0, stream>>>(c1, Wr2, br2, c2, 64, 128, 192, 192);
  conv3x3s2_kernel<<<(2 * 128 * 48 * 48) / 256, 256, 0, stream>>>(c2, Wr3, br3, pf, 128, 128, 96, 96);

  // Q/K/V projections (bf16 WMMA GEMMs); layouts make attention staging a memcpy:
  //   Q,K -> [b][h][n/m][d]   V -> [b][h][d][m]
  proj_qkv_kernel<<<2304, 128, 0, stream>>>(Wq, curr_frame, qb, 0, SCALE);
  proj_qkv_kernel<<<2304, 128, 0, stream>>>(Wk, prev_frame, kb, 0, 1.0f);
  const size_t vstride = (size_t)2 * 4 * N_ * DHEAD;
  proj_qkv_kernel<<<2304, 128, 0, stream>>>(Wvf, pf,       vb,               1, 1.0f);
  proj_qkv_kernel<<<2304, 128, 0, stream>>>(Wvn, prev_net, vb + vstride,     1, 1.0f);
  proj_qkv_kernel<<<2304, 128, 0, stream>>>(Wvi, prev_inp, vb + 2 * vstride, 1, 1.0f);

  // fused flash attention, 3 value streams share one softmax
  attn_kernel<<<8 * 18, 256, 0, stream>>>(qb, kb, vb, attb);

  // output projections fused with residual + gamma, straight into d_out
  const size_t astride = (size_t)2 * N_ * INNER;
  float* out = (float*)d_out;
  proj_out_kernel<<<576, 128, 0, stream>>>(attb,               Wpf, pf,       gamma_flow, out);
  proj_out_kernel<<<576, 128, 0, stream>>>(attb + astride,     Wpn, prev_net, gamma_net,  out + 1179648);
  proj_out_kernel<<<576, 128, 0, stream>>>(attb + 2 * astride, Wpi, prev_inp, gamma_inp,  out + 2 * 1179648);
}
